// DecoupledPointJAFAR_26319559590440
// MI455X (gfx1250) — matrix operand order repaired and checked by reading, not compiled, not run on hardware
//
#include <hip/hip_runtime.h>

// ---------- types ----------
typedef __attribute__((ext_vector_type(16))) _Float16 v16h;
typedef __attribute__((ext_vector_type(8)))  float    v8f;

struct alignas(16) H8 { _Float16 h[8]; };

#define DEV static __device__ __forceinline__

// ---------- wave32 cross-lane helpers (ds_swizzle, group-of-32 xor) ----------
template<int MASK> DEV float shxf(float v) {
  return __int_as_float(__builtin_amdgcn_ds_swizzle(__float_as_int(v), (MASK << 10) | 0x1f));
}
template<int MASK> DEV int shxi(int v) {
  return __builtin_amdgcn_ds_swizzle(v, (MASK << 10) | 0x1f);
}
DEV float red16sum(float v) { v += shxf<1>(v); v += shxf<2>(v); v += shxf<4>(v); v += shxf<8>(v); return v; }
DEV float red16max(float v) {
  v = fmaxf(v, shxf<1>(v)); v = fmaxf(v, shxf<2>(v));
  v = fmaxf(v, shxf<4>(v)); v = fmaxf(v, shxf<8>(v)); return v;
}

// ---------- fragment helpers ----------
DEV v16h load16(const _Float16* p) {
  const H8* q = (const H8*)p;
  H8 a = q[0], b = q[1];
  v16h r;
#pragma unroll
  for (int i = 0; i < 8; ++i) { r[i] = a.h[i]; r[i + 8] = b.h[i]; }
  return r;
}

DEV void zero4(v8f (&a)[4]) {
#pragma unroll
  for (int rb = 0; rb < 4; ++rb)
#pragma unroll
    for (int r = 0; r < 8; ++r) a[rb][r] = 0.f;
}

// out(64x16) += W(64x64) @ X(64x16); X in LDS point-major (pt*64+ch) f16,
// W pre-packed in A-fragment layout: frag[(rb*2+kb)*32 + lane][16 halves].
DEV void gemm64(const _Float16* ldsw, int nl, int hi, const _Float16* frag, int lane, v8f (&acc)[4]) {
  v16h b0 = load16(ldsw + nl * 64 + 16 * hi);        // k = 0..31  (lane half picks k range)
  v16h b1 = load16(ldsw + nl * 64 + 32 + 16 * hi);   // k = 32..63
#pragma unroll
  for (int rb = 0; rb < 4; ++rb) {
    v16h a0 = load16(frag + ((rb * 2 + 0) * 32 + lane) * 16);
    v16h a1 = load16(frag + ((rb * 2 + 1) * 32 + lane) * 16);
    acc[rb] = __builtin_amdgcn_wmma_f32_16x16x32_f16(false, a0, false, b0, (short)0, acc[rb], false, false);
    acc[rb] = __builtin_amdgcn_wmma_f32_16x16x32_f16(false, a1, false, b1, (short)0, acc[rb], false, false);
  }
}

DEV void add_bias(v8f (&acc)[4], const float* bias, int hi) {
#pragma unroll
  for (int rb = 0; rb < 4; ++rb)
#pragma unroll
    for (int r = 0; r < 8; ++r) acc[rb][r] += bias[rb * 16 + 8 * hi + r];
}

// C-layout tile (lane has rows rb*16+8*hi+r for column nl) -> f16 point-major store
DEV void store_tile_h(_Float16* dst /* base + pt*64 */, const v8f (&acc)[4], int hi) {
#pragma unroll
  for (int rb = 0; rb < 4; ++rb) {
    H8 h;
#pragma unroll
    for (int r = 0; r < 8; ++r) h.h[r] = (_Float16)acc[rb][r];
    *(H8*)(dst + rb * 16 + 8 * hi) = h;
  }
}

// 3->64 conv (+folded BN) + ReLU, write f16 point-major into wave LDS region
DEV void mlp3_relu(_Float16* lw, int nl, int hi, const float* W, const float* bias,
                   float x, float y, float z) {
#pragma unroll
  for (int j = 0; j < 4; ++j) {
    H8 h;
#pragma unroll
    for (int r = 0; r < 8; ++r) {
      int c = hi * 32 + j * 8 + r;
      float a = W[c * 3] * x + W[c * 3 + 1] * y + W[c * 3 + 2] * z + bias[c];
      h.h[r] = (_Float16)fmaxf(a, 0.f);
    }
    *(H8*)(lw + nl * 64 + hi * 32 + j * 8) = h;
  }
}

// ---------------------------------------------------------------------------
// K0: fold BN params, pack A-fragments for the five 64x64 matrices
// frag order: 0=ge_w2 1=q_w 2=k_w 3=bd_w1(BN-folded) 4=rp_w2
// cpar: [0..191] ge1 Wf  [192..255] ge1 bf  [256..447] rp1 Wf  [448..511] rp1 bf
//       [512..575] bd1 folded bias
// ---------------------------------------------------------------------------
__global__ void k0_prep(const float* ge_w1, const float* ge_b1, const float* ge_g1,
                        const float* ge_be1, const float* ge_m1, const float* ge_v1,
                        const float* ge_w2, const float* q_w, const float* k_w,
                        const float* bd_w1, const float* bd_b1, const float* bd_g,
                        const float* bd_be, const float* bd_m, const float* bd_v,
                        const float* rp_w1, const float* rp_b1, const float* rp_g,
                        const float* rp_be, const float* rp_m, const float* rp_v,
                        const float* rp_w2,
                        _Float16* frags, float* cpar) {
  int tid = threadIdx.x;
  if (tid < 64) {
    float s = ge_g1[tid] * rsqrtf(ge_v1[tid] + 1e-5f);
#pragma unroll
    for (int j = 0; j < 3; ++j) cpar[tid * 3 + j] = ge_w1[tid * 3 + j] * s;
    cpar[192 + tid] = s * (ge_b1[tid] - ge_m1[tid]) + ge_be1[tid];
    float sr = rp_g[tid] * rsqrtf(rp_v[tid] + 1e-5f);
#pragma unroll
    for (int j = 0; j < 3; ++j) cpar[256 + tid * 3 + j] = rp_w1[tid * 3 + j] * sr;
    cpar[448 + tid] = sr * (rp_b1[tid] - rp_m[tid]) + rp_be[tid];
    float sb = bd_g[tid] * rsqrtf(bd_v[tid] + 1e-5f);
    cpar[512 + tid] = sb * (bd_b1[tid] - bd_m[tid]) + bd_be[tid];
  }
  for (int idx = tid; idx < 5 * 4096; idx += blockDim.x) {
    int mat = idx >> 12, e = idx & 4095;
    int rbkb = e >> 9, lane = (e >> 4) & 31, v = e & 15;
    int rb = rbkb >> 1, kb = rbkb & 1;
    int m = rb * 16 + (lane & 15);
    int hi = lane >> 4;
    int k = kb * 32 + hi * 16 + v;  // contiguous k order (k = 16*hi + v), matches B loads
    const float* W;
    float scale = 1.0f;
    switch (mat) {
      case 0: W = ge_w2; break;
      case 1: W = q_w; break;
      case 2: W = k_w; break;
      case 3: W = bd_w1; scale = bd_g[m] * rsqrtf(bd_v[m] + 1e-5f); break;
      default: W = rp_w2; break;
    }
    frags[mat * 4096 + ((rb * 2 + kb) * 32 + lane) * 16 + v] = (_Float16)(W[m * 64 + k] * scale);
  }
}

// ---------------------------------------------------------------------------
// K1: HR pipeline — geom_enc -> Q (f16 point-major) + boundary head
// one wave per 16-point tile; 8 waves/block
// ---------------------------------------------------------------------------
__global__ void k1_hr(const float* xyz_hr, const float* cpar,
                      const _Float16* fragGE2, const _Float16* fragQ, const _Float16* fragBD1,
                      const float* ge_b2, const float* q_b, const float* bd_w2, const float* bd_b2,
                      _Float16* Q_ws, float* bdy_out, int N) {
  __shared__ alignas(16) _Float16 smem[8 * 1024];
  int lane = threadIdx.x & 31, widx = threadIdx.x >> 5;
  int t = blockIdx.x * 8 + widx;
  int n0 = (t * 16) % N, b = (t * 16) / N;
  int nl = lane & 15, hi = lane >> 4;
  int n = n0 + nl;
  _Float16* lw = smem + widx * 1024;

  float x = xyz_hr[(size_t)b * 3 * N + n];
  float y = xyz_hr[(size_t)b * 3 * N + N + n];
  float z = xyz_hr[(size_t)b * 3 * N + 2 * N + n];
  mlp3_relu(lw, nl, hi, cpar, cpar + 192, x, y, z);
  __syncthreads();

  v8f geom[4]; zero4(geom);
  gemm64(lw, nl, hi, fragGE2, lane, geom);
  add_bias(geom, ge_b2, hi);
  __syncthreads();
  store_tile_h(lw + nl * 64, geom, hi);   // geom_hr tile back to LDS, point-major
  __syncthreads();

  // Q = q_w @ geom + q_b
  v8f aq[4]; zero4(aq);
  gemm64(lw, nl, hi, fragQ, lane, aq);
  add_bias(aq, q_b, hi);
  store_tile_h(Q_ws + (size_t)(b * N + n) * 64, aq, hi);

  // boundary head: relu(bn(bd_w1 @ geom)) . bd_w2 -> sigmoid
  v8f ab[4]; zero4(ab);
  gemm64(lw, nl, hi, fragBD1, lane, ab);
  float partial = 0.f;
#pragma unroll
  for (int rb = 0; rb < 4; ++rb)
#pragma unroll
    for (int r = 0; r < 8; ++r) {
      int c = rb * 16 + 8 * hi + r;
      float h = fmaxf(ab[rb][r] + cpar[512 + c], 0.f);
      partial += bd_w2[c] * h;
    }
  partial += shxf<16>(partial);
  float prob = 1.f / (1.f + expf(-(partial + bd_b2[0])));
  if (hi == 0) bdy_out[(size_t)b * N + n] = prob;
}

// ---------------------------------------------------------------------------
// K2: LR pipeline — geom_enc, FiLM modulation, K (f16 point-major),
// packed (x,y,z,|q|^2) float4, val transpose to point-major f32
// ---------------------------------------------------------------------------
__global__ void k2_lr(const float* xyz_lr, const float* sft, const float* val_lr, const float* cpar,
                      const _Float16* fragGE2, const _Float16* fragK,
                      const float* ge_b2, const float* sc_w, const float* sc_b,
                      const float* sh_w, const float* sh_b, const float* k_b,
                      _Float16* K_ws, float* val_ws, float4* lrpts, int M) {
  __shared__ alignas(16) _Float16 smem[8 * 1024];
  int lane = threadIdx.x & 31, widx = threadIdx.x >> 5;
  int t = blockIdx.x * 8 + widx;
  int n0 = (t * 16) % M, b = (t * 16) / M;
  int nl = lane & 15, hi = lane >> 4;
  int n = n0 + nl;
  _Float16* lw = smem + widx * 1024;

  float x = xyz_lr[(size_t)b * 3 * M + n];
  float y = xyz_lr[(size_t)b * 3 * M + M + n];
  float z = xyz_lr[(size_t)b * 3 * M + 2 * M + n];
  mlp3_relu(lw, nl, hi, cpar, cpar + 192, x, y, z);
  __syncthreads();

  v8f geom[4]; zero4(geom);
  gemm64(lw, nl, hi, fragGE2, lane, geom);
  add_bias(geom, ge_b2, hi);
  __syncthreads();
  store_tile_h(lw + nl * 64, geom, hi);
  __syncthreads();

  // FiLM: geom * (scale+1) + shift, scale/shift = 6->64 convs of sft
  float sv[6];
#pragma unroll
  for (int jj = 0; jj < 6; ++jj) sv[jj] = sft[(size_t)b * 6 * M + (size_t)jj * M + n];
#pragma unroll
  for (int j = 0; j < 4; ++j) {
    H8 g = *(H8*)(lw + nl * 64 + hi * 32 + j * 8);
    H8 o;
#pragma unroll
    for (int r = 0; r < 8; ++r) {
      int c = hi * 32 + j * 8 + r;
      float sc = sc_b[c], sh = sh_b[c];
#pragma unroll
      for (int jj = 0; jj < 6; ++jj) { sc += sc_w[c * 6 + jj] * sv[jj]; sh += sh_w[c * 6 + jj] * sv[jj]; }
      o.h[r] = (_Float16)((float)g.h[r] * (sc + 1.f) + sh);
    }
    *(H8*)(lw + nl * 64 + hi * 32 + j * 8) = o;
  }
  __syncthreads();

  v8f ak[4]; zero4(ak);
  gemm64(lw, nl, hi, fragK, lane, ak);
  add_bias(ak, k_b, hi);
  store_tile_h(K_ws + (size_t)(b * M + n) * 64, ak, hi);

  if (hi == 0) {
    float4 q; q.x = x; q.y = y; q.z = z; q.w = x * x + y * y + z * z;
    lrpts[(size_t)b * M + n] = q;
  }
  // val transpose: (B,64,M) -> point-major (B,M,64)
#pragma unroll
  for (int j = 0; j < 8; ++j) {
    int c = hi * 32 + j * 4;
    float4 vv;
    vv.x = val_lr[(size_t)b * 64 * M + (size_t)(c + 0) * M + n];
    vv.y = val_lr[(size_t)b * 64 * M + (size_t)(c + 1) * M + n];
    vv.z = val_lr[(size_t)b * 64 * M + (size_t)(c + 2) * M + n];
    vv.w = val_lr[(size_t)b * 64 * M + (size_t)(c + 3) * M + n];
    *(float4*)&val_ws[(size_t)(b * M + n) * 64 + c] = vv;
  }
}

// ---------------------------------------------------------------------------
// K3: kNN — distances to LDS then 16 argmin passes (tie-break: lower index)
// one wave per HR point, 4 waves/block (64 KB LDS)
// ---------------------------------------------------------------------------
__global__ void k3_knn(const float* xyz_hr, const float4* lrpts, int* kidx, int N, int M) {
  __shared__ float dist[4][4096];
  int lane = threadIdx.x & 31, w = threadIdx.x >> 5;
  int p = blockIdx.x * 4 + w;
  int b = p / N, n = p % N;
  float px = xyz_hr[(size_t)b * 3 * N + n];
  float py = xyz_hr[(size_t)b * 3 * N + N + n];
  float pz = xyz_hr[(size_t)b * 3 * N + 2 * N + n];
  int cpl = M >> 5;
  for (int i = 0; i < cpl; ++i) {
    int m = lane + (i << 5);
    float4 q = lrpts[(size_t)b * M + m];
    dist[w][m] = q.w - 2.f * (px * q.x + py * q.y + pz * q.z);  // |p|^2 const: argmin-equivalent
  }
  __syncthreads();
  for (int pass = 0; pass < 16; ++pass) {
    float mv = 3.4e38f; int mi = 0x7fffffff;
    for (int i = 0; i < cpl; ++i) {
      int m = lane + (i << 5);
      float d = dist[w][m];
      if (d < mv) { mv = d; mi = m; }
    }
    { float ov = shxf<1>(mv);  int oi = shxi<1>(mi);  if (ov < mv || (ov == mv && oi < mi)) { mv = ov; mi = oi; } }
    { float ov = shxf<2>(mv);  int oi = shxi<2>(mi);  if (ov < mv || (ov == mv && oi < mi)) { mv = ov; mi = oi; } }
    { float ov = shxf<4>(mv);  int oi = shxi<4>(mi);  if (ov < mv || (ov == mv && oi < mi)) { mv = ov; mi = oi; } }
    { float ov = shxf<8>(mv);  int oi = shxi<8>(mi);  if (ov < mv || (ov == mv && oi < mi)) { mv = ov; mi = oi; } }
    { float ov = shxf<16>(mv); int oi = shxi<16>(mi); if (ov < mv || (ov == mv && oi < mi)) { mv = ov; mi = oi; } }
    if (lane == 0) { kidx[(size_t)p * 16 + pass] = mi; dist[w][mi] = 3.4e38f; }
    __syncthreads();
  }
}

// ---------------------------------------------------------------------------
// K4: attention — one wave per HR point; 16 neighbors are the WMMA tile cols
// ---------------------------------------------------------------------------
__global__ void k4_attn(const float* xyz_hr, const float* cpar, const _Float16* fragRP2,
                        const float* rp_b2, const int* kidx, const float4* lrpts,
                        const _Float16* K_ws, const _Float16* Q_ws, const float* val_ws,
                        float* out, int N, int M) {
  __shared__ alignas(16) _Float16 smem[8 * 1024];
  int lane = threadIdx.x & 31, widx = threadIdx.x >> 5;
  int p = blockIdx.x * 8 + widx;
  int b = p / N, n = p % N;
  int nl = lane & 15, hi = lane >> 4;
  _Float16* lw = smem + widx * 1024;

  int m = kidx[(size_t)p * 16 + nl];
  float px = xyz_hr[(size_t)b * 3 * N + n];
  float py = xyz_hr[(size_t)b * 3 * N + N + n];
  float pz = xyz_hr[(size_t)b * 3 * N + 2 * N + n];
  float4 q = lrpts[(size_t)b * M + m];

  mlp3_relu(lw, nl, hi, cpar + 256, cpar + 448, px - q.x, py - q.y, pz - q.z);
  __syncthreads();

  v8f acc[4]; zero4(acc);                       // pos_enc = rp_w2 @ pe
  gemm64(lw, nl, hi, fragRP2, lane, acc);

  const _Float16* kp = K_ws + (size_t)(b * M + m) * 64;
  const _Float16* qp = Q_ws + (size_t)(b * N + n) * 64;
  float partial = 0.f;
#pragma unroll
  for (int rb = 0; rb < 4; ++rb) {
    H8 kh = *(const H8*)(kp + rb * 16 + 8 * hi);
    H8 qh = *(const H8*)(qp + rb * 16 + 8 * hi);
#pragma unroll
    for (int r = 0; r < 8; ++r) {
      int c = rb * 16 + 8 * hi + r;
      float pe = acc[rb][r] + rp_b2[c];
      partial += (float)qh.h[r] * ((float)kh.h[r] + pe);
    }
  }
  partial += shxf<16>(partial);
  float logit = partial * 0.125f;               // 1/sqrt(64)
  float mx = red16max(logit);
  float e = expf(logit - mx);
  float s = red16sum(e);
  float w = e / s;

  const float* vp = val_ws + (size_t)(b * M + m) * 64;
#pragma unroll
  for (int c0 = 0; c0 < 64; c0 += 8) {
    int c = c0 + 4 * hi;
    float4 vv = *(const float4*)(vp + c);
    float ax = red16sum(w * vv.x);
    float ay = red16sum(w * vv.y);
    float az = red16sum(w * vv.z);
    float aw = red16sum(w * vv.w);
    if (nl == 0) {
      size_t base = (size_t)b * 64 * N + n;
      out[base + (size_t)(c + 0) * N] = ax;
      out[base + (size_t)(c + 1) * N] = ay;
      out[base + (size_t)(c + 2) * N] = az;
      out[base + (size_t)(c + 3) * N] = aw;
    }
  }
}

// ---------------------------------------------------------------------------
extern "C" void kernel_launch(void* const* d_in, const int* in_sizes, int n_in,
                              void* d_out, int out_size, void* d_ws, size_t ws_size,
                              hipStream_t stream) {
  (void)n_in; (void)out_size; (void)ws_size;
  const int B = 2;
  const int N = in_sizes[0] / 6;   // (B,3,N)
  const int M = in_sizes[1] / 6;   // (B,3,M)

  const float* xyz_hr = (const float*)d_in[0];
  const float* xyz_lr = (const float*)d_in[1];
  const float* sft    = (const float*)d_in[2];
  const float* val_lr = (const float*)d_in[3];
  const float* ge_w1  = (const float*)d_in[4];
  const float* ge_b1  = (const float*)d_in[5];
  const float* ge_g1  = (const float*)d_in[6];
  const float* ge_be1 = (const float*)d_in[7];
  const float* ge_m1  = (const float*)d_in[8];
  const float* ge_v1  = (const float*)d_in[9];
  const float* ge_w2  = (const float*)d_in[10];
  const float* ge_b2  = (const float*)d_in[11];
  const float* sc_w   = (const float*)d_in[12];
  const float* sc_b   = (const float*)d_in[13];
  const float* sh_w   = (const float*)d_in[14];
  const float* sh_b   = (const float*)d_in[15];
  const float* q_w    = (const float*)d_in[16];
  const float* q_b    = (const float*)d_in[17];
  const float* k_w    = (const float*)d_in[18];
  const float* k_b    = (const float*)d_in[19];
  const float* bd_w1  = (const float*)d_in[20];
  const float* bd_b1  = (const float*)d_in[21];
  const float* bd_g   = (const float*)d_in[22];
  const float* bd_be  = (const float*)d_in[23];
  const float* bd_m   = (const float*)d_in[24];
  const float* bd_v   = (const float*)d_in[25];
  const float* bd_w2  = (const float*)d_in[26];
  const float* bd_b2  = (const float*)d_in[27];
  const float* rp_w1  = (const float*)d_in[28];
  const float* rp_b1  = (const float*)d_in[29];
  const float* rp_g   = (const float*)d_in[30];
  const float* rp_be  = (const float*)d_in[31];
  const float* rp_m   = (const float*)d_in[32];
  const float* rp_v   = (const float*)d_in[33];
  const float* rp_w2  = (const float*)d_in[34];
  const float* rp_b2  = (const float*)d_in[35];

  char* ws = (char*)d_ws;
  _Float16* frags = (_Float16*)(ws);                  // 5 * 4096 halves = 40960 B
  float*    cpar  = (float*)(ws + 40960);             // 1024 floats
  _Float16* Qws   = (_Float16*)(ws + 45056);          // B*N*64 f16
  size_t oK = 45056 + (size_t)B * N * 64 * 2;
  _Float16* Kws   = (_Float16*)(ws + oK);             // B*M*64 f16
  size_t oV = oK + (size_t)B * M * 64 * 2;
  float*    Vws   = (float*)(ws + oV);                // B*M*64 f32
  size_t oL = oV + (size_t)B * M * 64 * 4;
  float4*   Lpts  = (float4*)(ws + oL);               // B*M float4
  size_t oI = oL + (size_t)B * M * 16;
  int*      Kidx  = (int*)(ws + oI);                  // B*N*16 int

  const _Float16* fragGE2 = frags;
  const _Float16* fragQ   = frags + 4096;
  const _Float16* fragK   = frags + 8192;
  const _Float16* fragBD1 = frags + 12288;
  const _Float16* fragRP2 = frags + 16384;

  float* out = (float*)d_out;
  float* bdy = out + (size_t)B * 64 * N;

  k0_prep<<<1, 256, 0, stream>>>(ge_w1, ge_b1, ge_g1, ge_be1, ge_m1, ge_v1,
                                 ge_w2, q_w, k_w,
                                 bd_w1, bd_b1, bd_g, bd_be, bd_m, bd_v,
                                 rp_w1, rp_b1, rp_g, rp_be, rp_m, rp_v, rp_w2,
                                 frags, cpar);

  k1_hr<<<dim3((unsigned)(B * N / 128)), 256, 0, stream>>>(
      xyz_hr, cpar, fragGE2, fragQ, fragBD1, ge_b2, q_b, bd_w2, bd_b2, Qws, bdy, N);

  k2_lr<<<dim3((unsigned)(B * M / 128)), 256, 0, stream>>>(
      xyz_lr, sft, val_lr, cpar, fragGE2, fragK, ge_b2,
      sc_w, sc_b, sh_w, sh_b, k_b, Kws, Vws, Lpts, M);

  k3_knn<<<dim3((unsigned)(B * N / 4)), 128, 0, stream>>>(xyz_hr, Lpts, Kidx, N, M);

  k4_attn<<<dim3((unsigned)(B * N / 8)), 256, 0, stream>>>(
      xyz_hr, cpar, fragRP2, rp_b2, Kidx, Lpts, Kws, Qws, Vws, out, N, M);
}
